// LossRecovery_59957743452648
// MI455X (gfx1250) — compile-verified, matrix operand order checked by reading, not run on hardware
//
#include <hip/hip_runtime.h>
#include <hip/hip_bf16.h>

// CDNA5 (gfx1250) bf16-WMMA implementation of the dual channel-attention block.
// All heavy GEMMs use V_WMMA_F32_16X16X32_BF16 (fp32 accumulate).
// Workspace requirement: ~276 MB (6x bf16 [16][4096][256] + 1x fp32 + small attn bufs).

typedef __attribute__((ext_vector_type(16))) __bf16 v16bf;
typedef __attribute__((ext_vector_type(8)))  __bf16 v8bf;
typedef __attribute__((ext_vector_type(8)))  float  v8f;

#define CH 256
#define HW 4096
#define NBL 16

// ---------------------------------------------------------------------------
// Fragment loader: 16x32 bf16 tile from "row-major, K-contiguous" memory.
// ISA A-layout (16-bit A 16x32): lane<16 -> row=lane, K {0..7} in elems 0..7,
// K {16..23} in elems 8..15; lane>=16 -> row=lane-16, K {8..15},{24..31}.
// The B operand (K x 16) uses the symmetric layout with lane holding column n,
// so loading B from an [N][K] row-major array uses the exact same code.
// ---------------------------------------------------------------------------
__device__ __forceinline__ v16bf load_frag(const __bf16* __restrict__ tile, int ld) {
  const int lane = threadIdx.x & 31;
  const int row  = lane & 15;
  const int k0   = (lane >> 4) << 3;  // 0 or 8
  const __bf16* p = tile + (size_t)row * ld + k0;
  union { v16bf v; v8bf h[2]; } u;
  u.h[0] = *(const v8bf*)(p);        // K = k0 .. k0+7      (16B aligned)
  u.h[1] = *(const v8bf*)(p + 16);   // K = k0+16 .. k0+23
  return u.v;
}

__device__ __forceinline__ v8f wmma_bf16(v16bf a, v16bf b, v8f c) {
  // (neg_a, A, neg_b, B, c_mod, C, reuse_a, reuse_b)
  return __builtin_amdgcn_wmma_f32_16x16x32_bf16(false, a, false, b, (short)0, c,
                                                 false, false);
}

// ---------------------------------------------------------------------------
// Packing kernels
// ---------------------------------------------------------------------------
__global__ void pack_x_kernel(const float* __restrict__ x, __bf16* __restrict__ xb) {
  const int row = blockIdx.x;                 // bl*4096 + s
  const int t   = threadIdx.x;                // channel 0..255
  xb[(size_t)row * CH + t] = (__bf16)x[(size_t)row * (CH + 3) + t];
}

__global__ void pack_mat_kernel(const float* __restrict__ src,
                                __bf16* __restrict__ dst, int n) {
  const int i = blockIdx.x * blockDim.x + threadIdx.x;
  if (i < n) dst[i] = (__bf16)src[i];
}

// v2[bl][d][s] = Vt[bl>>3*8 + d%8][s][(bl&7)*32 + d/8]  (stored [s][d]-major)
__global__ void pack_v2_kernel(const __bf16* __restrict__ Vt,
                               __bf16* __restrict__ V2) {
  __shared__ __bf16 buf[256];
  const int blk = blockIdx.x;                 // bl*4096 + s
  const int bl = blk >> 12, s = blk & 4095;
  const int b = bl >> 3, l = bl & 7;
  const int t = threadIdx.x;
  const int lp = t >> 5;                      // source layer 0..7
  const int j  = t & 31;                      // 0..31 (coalesced 64B per wave)
  buf[j * 8 + lp] = Vt[((size_t)(b * 8 + lp) * HW + s) * CH + l * 32 + j];
  __syncthreads();
  V2[(size_t)blk * CH + t] = buf[t];
}

// ---------------------------------------------------------------------------
// 1x1 conv: Y = X * W^T + b  (M=4096 pixels, N=256 out-ch, K=256 in-ch)
// mode 0: store natural [s][d] bf16
// mode 1: store transposed [d][s] bf16 (LDS transpose, 128-bit stores)
// mode 2: like 1 but column index spatially permuted s' = (s%64)*64 + s/64
// ---------------------------------------------------------------------------
__global__ __launch_bounds__(256) void conv_gemm_kernel(
    const __bf16* __restrict__ X,   // [16][4096][256]
    const __bf16* __restrict__ W,   // [256][256] (d-major, c-contiguous)
    const float*  __restrict__ bias,
    __bf16* __restrict__ Y, int mode) {
  __shared__ float lds[8][16][17];
  const int bl   = blockIdx.z;
  const int wave = threadIdx.x >> 5;
  const int lane = threadIdx.x & 31;
  const int s0 = blockIdx.x * 128 + (wave & 3) * 32;
  const int d0 = blockIdx.y * 64  + (wave >> 2) * 32;
  const __bf16* Xb = X + (size_t)bl * HW * CH;

  v8f acc[2][2] = {};
  for (int kk = 0; kk < CH; kk += 32) {
    v16bf a0 = load_frag(Xb + (size_t)s0 * CH + kk, CH);
    v16bf a1 = load_frag(Xb + (size_t)(s0 + 16) * CH + kk, CH);
    v16bf b0 = load_frag(W + (size_t)d0 * CH + kk, CH);
    v16bf b1 = load_frag(W + (size_t)(d0 + 16) * CH + kk, CH);
    acc[0][0] = wmma_bf16(a0, b0, acc[0][0]);
    acc[0][1] = wmma_bf16(a0, b1, acc[0][1]);
    acc[1][0] = wmma_bf16(a1, b0, acc[1][0]);
    acc[1][1] = wmma_bf16(a1, b1, acc[1][1]);
  }

  const int n = lane & 15, mb = (lane >> 4) << 3;   // D layout: (m=mb+r, n)
  __bf16* Yb = Y + (size_t)bl * HW * CH;
  for (int ti = 0; ti < 2; ++ti)
    for (int tj = 0; tj < 2; ++tj) {
      const int st = s0 + ti * 16, dt = d0 + tj * 16;
      const float bv = bias[dt + n];
      v8f a = acc[ti][tj];
      if (mode == 0) {
        for (int r = 0; r < 8; ++r)
          Yb[(size_t)(st + mb + r) * CH + dt + n] = (__bf16)(a[r] + bv);
      } else {
        for (int r = 0; r < 8; ++r) lds[wave][mb + r][n] = a[r] + bv;
        __syncthreads();
        const int dp = lane & 15, sb = (lane >> 4) << 3;
        if (mode == 1) {
          union { v8bf v; __bf16 e[8]; } o;
          for (int j = 0; j < 8; ++j) o.e[j] = (__bf16)lds[wave][sb + j][dp];
          *(v8bf*)(Yb + (size_t)(dt + dp) * HW + st + sb) = o.v;
        } else {
          for (int j = 0; j < 8; ++j) {
            const int s = st + sb + j;
            const int sp = ((s & 63) << 6) | (s >> 6);   // (h,w) -> (w,h)
            Yb[(size_t)(dt + dp) * HW + sp] = (__bf16)lds[wave][sb + j][dp];
          }
        }
        __syncthreads();
      }
    }
}

// ---------------------------------------------------------------------------
// Attention logits: L[c][d] = sum_s Q[c][s] * K[d][s]   (M=N=256, K=4096)
// ---------------------------------------------------------------------------
__global__ __launch_bounds__(256) void logits_gemm_kernel(
    const __bf16* __restrict__ Q,   // [16][256][4096]
    const __bf16* __restrict__ K,   // [16][256][4096]
    float* __restrict__ Lg) {       // [16][256][256]
  const int bl   = blockIdx.z;
  const int wave = threadIdx.x >> 5;
  const int lane = threadIdx.x & 31;
  const int c0 = blockIdx.x * 128 + (wave & 3) * 32;
  const int d0 = blockIdx.y * 64  + (wave >> 2) * 32;
  const __bf16* Qb = Q + (size_t)bl * CH * HW;
  const __bf16* Kb = K + (size_t)bl * CH * HW;

  v8f acc[2][2] = {};
  for (int kk = 0; kk < HW; kk += 32) {
    __builtin_prefetch(Qb + (size_t)c0 * HW + kk + 256, 0, 1);
    v16bf a0 = load_frag(Qb + (size_t)c0 * HW + kk, HW);
    v16bf a1 = load_frag(Qb + (size_t)(c0 + 16) * HW + kk, HW);
    v16bf b0 = load_frag(Kb + (size_t)d0 * HW + kk, HW);
    v16bf b1 = load_frag(Kb + (size_t)(d0 + 16) * HW + kk, HW);
    acc[0][0] = wmma_bf16(a0, b0, acc[0][0]);
    acc[0][1] = wmma_bf16(a0, b1, acc[0][1]);
    acc[1][0] = wmma_bf16(a1, b0, acc[1][0]);
    acc[1][1] = wmma_bf16(a1, b1, acc[1][1]);
  }
  const int n = lane & 15, mb = (lane >> 4) << 3;
  float* Lb = Lg + (size_t)bl * CH * CH;
  for (int ti = 0; ti < 2; ++ti)
    for (int tj = 0; tj < 2; ++tj) {
      v8f a = acc[ti][tj];
      for (int r = 0; r < 8; ++r)
        Lb[(size_t)(c0 + ti * 16 + mb + r) * CH + d0 + tj * 16 + n] = a[r];
    }
}

// ---------------------------------------------------------------------------
// Row softmax over 256 logits (one wave per row), output bf16.
// ---------------------------------------------------------------------------
__global__ __launch_bounds__(256) void softmax_kernel(
    const float* __restrict__ Lg, __bf16* __restrict__ A) {
  const int wave = threadIdx.x >> 5;
  const int lane = threadIdx.x & 31;
  const int row  = blockIdx.x * 8 + wave;     // 0 .. 16*256-1
  const float* p = Lg + (size_t)row * CH + lane * 8;
  float v[8];
  float m = -3.0e38f;
  for (int j = 0; j < 8; ++j) { v[j] = p[j]; m = fmaxf(m, v[j]); }
  for (int off = 16; off > 0; off >>= 1) m = fmaxf(m, __shfl_xor(m, off, 32));
  float ssum = 0.f;
  for (int j = 0; j < 8; ++j) { v[j] = __expf(v[j] - m); ssum += v[j]; }
  for (int off = 16; off > 0; off >>= 1) ssum += __shfl_xor(ssum, off, 32);
  const float inv = 1.0f / ssum;
  union { v8bf o; __bf16 e[8]; } u;
  for (int j = 0; j < 8; ++j) u.e[j] = (__bf16)(v[j] * inv);
  *(v8bf*)(A + (size_t)row * CH + lane * 8) = u.o;
}

// ---------------------------------------------------------------------------
// out[s][c] = sum_d attn[c][d] * V[s][d]; fused epilogue:
//   res = gamma * out + resid[(bl*4096+s)*rstride + c]
//   write res to outF (fp32) and optionally bf16 copy to outB.
// ---------------------------------------------------------------------------
__global__ __launch_bounds__(256) void apply_attn_kernel(
    const __bf16* __restrict__ A,     // [16][256][256]
    const __bf16* __restrict__ V,     // [16][4096][256]
    const float*  __restrict__ resid, // pixel-row stride = rstride (259 or 256)
    int rstride,
    const float*  __restrict__ gamma, // 1 element
    float*  __restrict__ outF,        // [16][4096][256] fp32
    __bf16* __restrict__ outB) {      // nullable bf16 copy
  __shared__ float lds[8][16][17];
  const int bl   = blockIdx.z;
  const int wave = threadIdx.x >> 5;
  const int lane = threadIdx.x & 31;
  const int c0 = blockIdx.x * 64  + (wave >> 2) * 32;
  const int s0 = blockIdx.y * 128 + (wave & 3) * 32;
  const __bf16* Ab = A + (size_t)bl * CH * CH;
  const __bf16* Vb = V + (size_t)bl * HW * CH;

  v8f acc[2][2] = {};
  for (int kk = 0; kk < CH; kk += 32) {
    v16bf a0 = load_frag(Ab + (size_t)c0 * CH + kk, CH);
    v16bf a1 = load_frag(Ab + (size_t)(c0 + 16) * CH + kk, CH);
    v16bf b0 = load_frag(Vb + (size_t)s0 * CH + kk, CH);
    v16bf b1 = load_frag(Vb + (size_t)(s0 + 16) * CH + kk, CH);
    acc[0][0] = wmma_bf16(a0, b0, acc[0][0]);
    acc[0][1] = wmma_bf16(a0, b1, acc[0][1]);
    acc[1][0] = wmma_bf16(a1, b0, acc[1][0]);
    acc[1][1] = wmma_bf16(a1, b1, acc[1][1]);
  }

  const float g = gamma[0];
  const int n = lane & 15, mb = (lane >> 4) << 3;
  for (int ti = 0; ti < 2; ++ti)        // c sub-tile
    for (int tj = 0; tj < 2; ++tj) {    // s sub-tile
      const int ct = c0 + ti * 16, st = s0 + tj * 16;
      v8f a = acc[ti][tj];
      for (int r = 0; r < 8; ++r) lds[wave][mb + r][n] = a[r];  // [c'][s']
      __syncthreads();
      const int sp = lane & 15, cb = (lane >> 4) << 3;
      const int s = st + sp;
      const size_t prow = (size_t)bl * HW + s;
      const float* rp = resid + prow * rstride + ct + cb;
      float o[8];
      for (int j = 0; j < 8; ++j) o[j] = g * lds[wave][cb + j][sp] + rp[j];
      union { v8f v; float e[8]; } uf;
      for (int j = 0; j < 8; ++j) uf.e[j] = o[j];
      *(v8f*)(outF + prow * CH + ct + cb) = uf.v;
      if (outB) {
        union { v8bf v; __bf16 e[8]; } ub;
        for (int j = 0; j < 8; ++j) ub.e[j] = (__bf16)o[j];
        *(v8bf*)(outB + prow * CH + ct + cb) = ub.v;
      }
      __syncthreads();
    }
}

// ---------------------------------------------------------------------------
extern "C" void kernel_launch(void* const* d_in, const int* in_sizes, int n_in,
                              void* d_out, int out_size, void* d_ws, size_t ws_size,
                              hipStream_t stream) {
  (void)in_sizes; (void)n_in; (void)out_size; (void)ws_size;
  const float* x     = (const float*)d_in[0];
  // d_in[1]=mask, d_in[2]=spatial_correction_matrix: unused by the reference.
  const float* w_sq = (const float*)d_in[3];  const float* b_sq = (const float*)d_in[4];
  const float* w_sk = (const float*)d_in[5];  const float* b_sk = (const float*)d_in[6];
  const float* w_sv = (const float*)d_in[7];  const float* b_sv = (const float*)d_in[8];
  const float* w_tq = (const float*)d_in[9];  const float* b_tq = (const float*)d_in[10];
  const float* w_tk = (const float*)d_in[11]; const float* b_tk = (const float*)d_in[12];
  const float* w_tv = (const float*)d_in[13]; const float* b_tv = (const float*)d_in[14];
  const float* s_gamma = (const float*)d_in[15];
  const float* t_gamma = (const float*)d_in[16];

  const size_t NE = (size_t)NBL * HW * CH;     // 16,777,216 elements
  char* p = (char*)d_ws;
  __bf16* Xbf  = (__bf16*)p; p += NE * 2;      // x_s in bf16 [bl][s][c]
  __bf16* Qt   = (__bf16*)p; p += NE * 2;      // q / q2, [bl][c][s]
  __bf16* Kt   = (__bf16*)p; p += NE * 2;      // k / k2, [bl][c][s]
  __bf16* Vb   = (__bf16*)p; p += NE * 2;      // v / tv conv, [bl][s][d]
  __bf16* V2   = (__bf16*)p; p += NE * 2;      // gathered v2, [bl][s][d]
  __bf16* X1b  = (__bf16*)p; p += NE * 2;      // x1 bf16
  float*  X1f  = (float*)p;  p += NE * 4;      // x1 fp32
  float*  Lg   = (float*)p;  p += (size_t)NBL * CH * CH * 4;
  __bf16* Attn = (__bf16*)p; p += (size_t)NBL * CH * CH * 2;
  __bf16* Wbf  = (__bf16*)p; p += (size_t)6 * CH * CH * 2;

  const dim3 blk(256);
  const dim3 gconv(HW / 128, CH / 64, NBL);    // (32,4,16)
  const dim3 glog(CH / 128, CH / 64, NBL);     // (2,4,16)
  const dim3 gapp(CH / 64, HW / 128, NBL);     // (4,32,16)

  // pack weights & activations to bf16
  const float* ws6[6] = {w_sq, w_sk, w_sv, w_tq, w_tk, w_tv};
  for (int i = 0; i < 6; ++i)
    pack_mat_kernel<<<(CH * CH + 255) / 256, blk, 0, stream>>>(
        ws6[i], Wbf + (size_t)i * CH * CH, CH * CH);
  pack_x_kernel<<<NBL * HW, blk, 0, stream>>>(x, Xbf);

  // ---- spatial attention ----
  conv_gemm_kernel<<<gconv, blk, 0, stream>>>(Xbf, Wbf + 0 * CH * CH, b_sq, Qt, 1);
  conv_gemm_kernel<<<gconv, blk, 0, stream>>>(Xbf, Wbf + 1 * CH * CH, b_sk, Kt, 1);
  conv_gemm_kernel<<<gconv, blk, 0, stream>>>(Xbf, Wbf + 2 * CH * CH, b_sv, Vb, 0);
  logits_gemm_kernel<<<glog, blk, 0, stream>>>(Qt, Kt, Lg);
  softmax_kernel<<<NBL * CH / 8, blk, 0, stream>>>(Lg, Attn);
  apply_attn_kernel<<<gapp, blk, 0, stream>>>(Attn, Vb, x, CH + 3, s_gamma,
                                              X1f, X1b);

  // ---- temporal attention ----
  conv_gemm_kernel<<<gconv, blk, 0, stream>>>(X1b, Wbf + 3 * CH * CH, b_tq, Qt, 1);
  conv_gemm_kernel<<<gconv, blk, 0, stream>>>(Xbf, Wbf + 4 * CH * CH, b_tk, Kt, 2);
  conv_gemm_kernel<<<gconv, blk, 0, stream>>>(Xbf, Wbf + 5 * CH * CH, b_tv, Vb, 0);
  pack_v2_kernel<<<NBL * HW, blk, 0, stream>>>(Vb, V2);
  logits_gemm_kernel<<<glog, blk, 0, stream>>>(Qt, Kt, Lg);
  softmax_kernel<<<NBL * CH / 8, blk, 0, stream>>>(Lg, Attn);
  apply_attn_kernel<<<gapp, blk, 0, stream>>>(Attn, V2, X1f, CH, t_gamma,
                                              (float*)d_out, nullptr);
}
